// ImageEncoderViT_46858093199370
// MI455X (gfx1250) — compile-verified, matrix operand order checked by reading, not run on hardware
//
#include <hip/hip_runtime.h>
#include <hip/hip_bf16.h>

typedef __attribute__((ext_vector_type(16))) _Float16 v16h;
typedef __attribute__((ext_vector_type(8)))  float    v8f;
typedef __attribute__((ext_vector_type(4)))  _Float16 v4h;
typedef __attribute__((ext_vector_type(2)))  int      v2i;

#define GF_GELU 1

#if defined(__has_builtin)
#if __has_builtin(__builtin_amdgcn_global_load_async_to_lds_b64)
#define HAVE_ASYNC_LDS 1
#endif
#endif
#ifndef HAVE_ASYNC_LDS
#define HAVE_ASYNC_LDS 0
#endif

// 8-byte global -> LDS copy; async (no VGPR round-trip, ASYNCcnt-tracked) when
// the toolchain exposes the gfx1250 builtin, else synchronous vector copy.
// Builtin signature (from toolchain diagnostics): (v2i AS1*, v2i AS3*, Ii, Ii).
__device__ __forceinline__ void copy_b64_to_lds(void* lds, const void* g) {
#if HAVE_ASYNC_LDS
    __builtin_amdgcn_global_load_async_to_lds_b64(
        (__attribute__((address_space(1))) v2i*)g,
        (__attribute__((address_space(3))) v2i*)lds, 0, 0);
#else
    *(v4h*)lds = *(const v4h*)g;
#endif
}

__device__ __forceinline__ void wait_async_lds() {
#if HAVE_ASYNC_LDS
#if __has_builtin(__builtin_amdgcn_s_wait_asynccnt)
    __builtin_amdgcn_s_wait_asynccnt(0);
#else
    asm volatile("s_wait_asynccnt 0x0" ::: "memory");
#endif
#endif
}

// ---------------------------------------------------------------------------
// Stage a 64(rows) x 64(k) f16 tile into LDS [64][68].
// Fast path: 8 async b64 copies per thread. Tail path: branchless clamped
// synchronous loads (select 0 for OOB) - no exec-mask branching per element.
// ---------------------------------------------------------------------------
__device__ __forceinline__
void stage_tile(_Float16 (*__restrict__ dst)[68], const _Float16* __restrict__ src,
                int ld, int row0, int k0, int R, int K, int tid)
{
    const int r  = tid >> 1;             // 0..63
    const int c0 = (tid & 1) * 32;       // 0 or 32
    const int gr = row0 + r;
    if ((k0 + 64 <= K) && (gr < R)) {
        const _Float16* s = src + (size_t)gr * ld + k0 + c0;
#pragma unroll
        for (int i = 0; i < 8; ++i)
            copy_b64_to_lds(&dst[r][c0 + i * 4], s + i * 4);
    } else {
#pragma unroll
        for (int c = 0; c < 32; ++c) {
            int gk = k0 + c0 + c;
            bool ok = (gr < R) && (gk < K);
            size_t idx = ok ? ((size_t)gr * ld + gk) : 0;
            _Float16 v = src[idx];
            dst[r][c0 + c] = ok ? v : (_Float16)0.0f;
        }
    }
}

// ---------------------------------------------------------------------------
// Generic batched f16 WMMA GEMM:  C[z] = epilogue(A[z] (MxK) @ B[z])
// B is stored TRANSPOSED: Bt [N][K] row-major (ldb = k-stride).
// 128 threads = 4 waves; block computes a 64x64 tile; K-step 64 (2 WMMA
// sub-steps => 8 v_wmma per step); double-buffered LDS with async fills so the
// next tile streams into LDS while the current one is consumed.
// ---------------------------------------------------------------------------
__global__ __launch_bounds__(128)
void k_gemm(const _Float16* __restrict__ A, int lda, size_t sA,
            const _Float16* __restrict__ Bt, int ldb, size_t sB,
            float* __restrict__ C, int ldc, size_t sC,
            _Float16* __restrict__ Ch, int ldch, size_t sCh,
            const float* __restrict__ bias,
            const float* __restrict__ res, int ldres, size_t sRes,
            int M, int N, int K, int flags)
{
    __shared__ _Float16 As[2][64][68];   // [buf][m][k]
    __shared__ _Float16 Bs[2][64][68];   // [buf][n][k]

    const int z = blockIdx.z;
    A  += (size_t)z * sA;
    Bt += (size_t)z * sB;
    if (C)   C   += (size_t)z * sC;
    if (Ch)  Ch  += (size_t)z * sCh;
    if (res) res += (size_t)z * sRes;

    const int m0 = blockIdx.y * 64;
    const int n0 = blockIdx.x * 64;
    const int tid  = threadIdx.x;
    const int lane = tid & 31;
    const int wave = tid >> 5;
    const int lrow = lane & 15;          // M row (A) / N col (B) within 16
    const int kgrp = (lane >> 4) * 8;    // lanes 0-15: k 0..7, lanes 16-31: k 8..15

    v8f zero = {0.f,0.f,0.f,0.f,0.f,0.f,0.f,0.f};
    v8f acc[4];
#pragma unroll
    for (int j = 0; j < 4; ++j) acc[j] = zero;

    // prologue: fill buffer 0
    stage_tile(As[0], A,  lda, m0, 0, M, K, tid);
    stage_tile(Bs[0], Bt, ldb, n0, 0, N, K, tid);
    wait_async_lds();
    __syncthreads();

    int cur = 0;
    for (int k0 = 0; k0 < K; k0 += 64) {
        const int nxt = cur ^ 1;
        // stream next K tile into the other buffer while we compute
        if (k0 + 64 < K) {
            stage_tile(As[nxt], A,  lda, m0, k0 + 64, M, K, tid);
            stage_tile(Bs[nxt], Bt, ldb, n0, k0 + 64, N, K, tid);
        }

        _Float16 (*__restrict__ Ac)[68] = As[cur];
        _Float16 (*__restrict__ Bc)[68] = Bs[cur];
#pragma unroll
        for (int ks = 0; ks < 64; ks += 32) {
            v16h a;
#pragma unroll
            for (int e = 0; e < 8; ++e) {
                a[e]     = Ac[wave * 16 + lrow][ks + kgrp + e];
                a[e + 8] = Ac[wave * 16 + lrow][ks + 16 + kgrp + e];
            }
#pragma unroll
            for (int j = 0; j < 4; ++j) {
                v16h b;
#pragma unroll
                for (int e = 0; e < 8; ++e) {
                    b[e]     = Bc[j * 16 + lrow][ks + kgrp + e];
                    b[e + 8] = Bc[j * 16 + lrow][ks + 16 + kgrp + e];
                }
                acc[j] = __builtin_amdgcn_wmma_f32_16x16x32_f16(
                    false, a, false, b, (short)0, acc[j], false, false);
            }
        }

        wait_async_lds();     // this wave's async fills of `nxt` complete
        __syncthreads();      // everyone done reading `cur` / writing `nxt`
        cur = nxt;
    }

    // epilogue: C/D layout -> VGPR v holds (row = v + 8*(lane>=16), col = lane&15)
    const int rowOff = (lane >> 4) * 8;
    const int col    = lane & 15;
    auto epi = [&](bool guard) {
#pragma unroll
        for (int j = 0; j < 4; ++j) {
#pragma unroll
            for (int v = 0; v < 8; ++v) {
                int gm = m0 + wave * 16 + rowOff + v;
                int gn = n0 + j * 16 + col;
                if (!guard || (gm < M && gn < N)) {
                    float val = acc[j][v];
                    if (bias) val += bias[gn];
                    if (flags & GF_GELU) val = 0.5f * val * (1.0f + erff(val * 0.70710678118f));
                    if (res)  val += res[(size_t)gm * ldres + gn];
                    if (C)  C [(size_t)gm * ldc  + gn] = val;
                    if (Ch) Ch[(size_t)gm * ldch + gn] = (_Float16)val;
                }
            }
        }
    };
    if ((m0 + 64 <= M) && (n0 + 64 <= N)) epi(false); else epi(true);
}

// ---------------------------------------------------------------------------
// Small helper kernels
// ---------------------------------------------------------------------------
__global__ void k_cvt(const float* __restrict__ s, _Float16* __restrict__ d, int n) {
    int i = blockIdx.x * blockDim.x + threadIdx.x;
    if (i < n) d[i] = (_Float16)s[i];
}

// batched transpose-convert: src [z][R][C] f32 -> dst [z][C][R] f16
__global__ void k_cvt_t(const float* __restrict__ s, _Float16* __restrict__ d, int R, int C) {
    size_t bo = (size_t)blockIdx.y * R * C;
    int i = blockIdx.x * blockDim.x + threadIdx.x;
    if (i >= R * C) return;
    int r = i / C, c = i % C;
    d[bo + (size_t)c * R + r] = (_Float16)s[bo + i];
}

// patch-embed im2col: x (3,512,512) -> Xp [1024][768], f = c*256 + iy*16 + ix
__global__ void k_im2col_patch(const float* __restrict__ x, _Float16* __restrict__ Xp) {
    int i = blockIdx.x * blockDim.x + threadIdx.x;
    if (i >= 1024 * 768) return;
    int f = i % 768, t = i / 768;
    int c = f / 256, rem = f % 256, iy = rem / 16, ix = rem % 16;
    int py = t / 32, px = t % 32;
    Xp[i] = (_Float16)x[(size_t)c * 512 * 512 + (size_t)(py * 16 + iy) * 512 + (px * 16 + ix)];
}

// LayerNorm over last dim (block per row, 256 threads). yh optional f16 mirror.
__global__ void k_layernorm(const float* __restrict__ x, const float* __restrict__ w,
                            const float* __restrict__ b, float* __restrict__ y,
                            _Float16* __restrict__ yh, int C) {
    __shared__ float red[256];
    int row = blockIdx.x;
    const float* xr = x + (size_t)row * C;
    int tid = threadIdx.x;
    float s = 0.f;
    for (int c = tid; c < C; c += 256) s += xr[c];
    red[tid] = s; __syncthreads();
    for (int st = 128; st > 0; st >>= 1) { if (tid < st) red[tid] += red[tid + st]; __syncthreads(); }
    float mean = red[0] / C; __syncthreads();
    float v = 0.f;
    for (int c = tid; c < C; c += 256) { float d = xr[c] - mean; v += d * d; }
    red[tid] = v; __syncthreads();
    for (int st = 128; st > 0; st >>= 1) { if (tid < st) red[tid] += red[tid + st]; __syncthreads(); }
    float rstd = rsqrtf(red[0] / C + 1e-6f); __syncthreads();
    for (int c = tid; c < C; c += 256) {
        float val = (xr[c] - mean) * rstd * w[c] + b[c];
        if (y)  y [(size_t)row * C + c] = val;
        if (yh) yh[(size_t)row * C + c] = (_Float16)val;
    }
}

// Global attention extract: qkv [1024][2304] ->
//   q [12][1024][64] (scaled), k [12][1024][64], vT [12][64][1024]  (f16)
__global__ void k_extract_global(const float* __restrict__ qkv,
                                 _Float16* __restrict__ q, _Float16* __restrict__ k,
                                 _Float16* __restrict__ vT, float scale) {
    int i = blockIdx.x * blockDim.x + threadIdx.x;
    if (i >= 12 * 1024 * 64) return;
    int d = i & 63;
    int t = (i >> 6) & 1023;
    int h = i >> 16;
    const float* base = qkv + (size_t)t * 2304 + h * 64 + d;
    q [((size_t)h * 1024 + t) * 64 + d] = (_Float16)(base[0] * scale);
    k [((size_t)h * 1024 + t) * 64 + d] = (_Float16)base[768];
    vT[(size_t)h * 65536 + (size_t)d * 1024 + t] = (_Float16)base[1536];
}

// Window extract (9 windows of 14x14 over padded 42x42; pad tokens = qkv bias)
//   q [108][196][64], k [108][196][64], vT [108][64][196]
__global__ void k_extract_window(const float* __restrict__ qkv, const float* __restrict__ qkvb,
                                 _Float16* __restrict__ q, _Float16* __restrict__ k,
                                 _Float16* __restrict__ vT, float scale) {
    int i = blockIdx.x * blockDim.x + threadIdx.x;
    if (i >= 108 * 196 * 64) return;
    int d  = i & 63;
    int lt = (i >> 6) % 196;
    int bh = (i >> 6) / 196;         // win*12 + head
    int head = bh % 12, win = bh / 12;
    int ly = lt / 14, lx = lt % 14;
    int gy = (win / 3) * 14 + ly, gx = (win % 3) * 14 + lx;
    int cq = head * 64 + d;
    float qv, kv, vv;
    if (gy < 32 && gx < 32) {
        const float* base = qkv + (size_t)(gy * 32 + gx) * 2304 + cq;
        qv = base[0]; kv = base[768]; vv = base[1536];
    } else {
        qv = qkvb[cq]; kv = qkvb[768 + cq]; vv = qkvb[1536 + cq];
    }
    q [((size_t)bh * 196 + lt) * 64 + d] = (_Float16)(qv * scale);
    k [((size_t)bh * 196 + lt) * 64 + d] = (_Float16)kv;
    vT[(size_t)bh * (64 * 196) + (size_t)d * 196 + lt] = (_Float16)vv;
}

// Decomposed rel-pos: rel[b][qh][qw][k] = mul * sum_d q[b][qh][qw][d]*R[qi-k+S-1][d]
// (q was pre-scaled by 0.125 for the QK GEMM, so mul=8 restores the unscaled q)
__global__ void k_relpos(const _Float16* __restrict__ q, const float* __restrict__ R,
                         float* __restrict__ rel, int H, int W, int axis, float mul) {
    int S = axis ? W : H;
    int b = blockIdx.y;
    int o = blockIdx.x * blockDim.x + threadIdx.x;
    if (o >= H * W * S) return;
    int k  = o % S;
    int qw = (o / S) % W;
    int qh = o / (S * W);
    int qi = axis ? qw : qh;
    const _Float16* qp = q + ((size_t)b * H * W + qh * W + qw) * 64;
    const float*    rp = R + (size_t)(qi - k + S - 1) * 64;
    float s = 0.f;
    for (int d = 0; d < 64; ++d) s += (float)qp[d] * rp[d];
    rel[(size_t)b * (H * W * S) + o] = s * mul;
}

// attn[b][q][k] += rel_h[b][q][k/W] + rel_w[b][q][k%W]; softmax over k; write f16
__global__ void k_rel_softmax(float* __restrict__ attn, _Float16* __restrict__ attn_h,
                              const float* __restrict__ relh, const float* __restrict__ relw,
                              int H, int W) {
    __shared__ float red[256];
    int HW = H * W;
    int b = blockIdx.y, q = blockIdx.x;
    int tid = threadIdx.x;
    size_t rowOff = ((size_t)b * HW + q) * HW;
    float* row = attn + rowOff;
    const float* rh = relh + ((size_t)b * HW + q) * H;
    const float* rw = relw + ((size_t)b * HW + q) * W;
    float mx = -1e30f;
    for (int k = tid; k < HW; k += 256) {
        float v = row[k] + rh[k / W] + rw[k % W];
        row[k] = v;
        mx = fmaxf(mx, v);
    }
    red[tid] = mx; __syncthreads();
    for (int st = 128; st > 0; st >>= 1) { if (tid < st) red[tid] = fmaxf(red[tid], red[tid + st]); __syncthreads(); }
    mx = red[0]; __syncthreads();
    float s = 0.f;
    for (int k = tid; k < HW; k += 256) {
        float e = __expf(row[k] - mx);
        row[k] = e;
        s += e;
    }
    red[tid] = s; __syncthreads();
    for (int st = 128; st > 0; st >>= 1) { if (tid < st) red[tid] += red[tid + st]; __syncthreads(); }
    float inv = 1.f / red[0]; __syncthreads();
    for (int k = tid; k < HW; k += 256)
        attn_h[rowOff + k] = (_Float16)(row[k] * inv);
}

// window unpartition of attn-output heads into [token][head*64+d] (f32 + f16)
__global__ void k_win_scatter(const float* __restrict__ wout, float* __restrict__ out,
                              _Float16* __restrict__ out_h) {
    int i = blockIdx.x * blockDim.x + threadIdx.x;
    if (i >= 108 * 196 * 64) return;
    int d  = i & 63;
    int lt = (i >> 6) % 196;
    int bh = (i >> 6) / 196;
    int head = bh % 12, win = bh / 12;
    int ly = lt / 14, lx = lt % 14;
    int gy = (win / 3) * 14 + ly, gx = (win % 3) * 14 + lx;
    if (gy >= 32 || gx >= 32) return;
    float v = wout[i];
    size_t o = (size_t)(gy * 32 + gx) * 768 + head * 64 + d;
    out[o]   = v;
    out_h[o] = (_Float16)v;
}

// neck conv2 im2col: t2 [1024][256] (NHWC) -> col [1024][2304], f = c*9 + dy*3 + dx
__global__ void k_im2col_3x3(const float* __restrict__ t2, _Float16* __restrict__ col) {
    int i = blockIdx.x * blockDim.x + threadIdx.x;
    if (i >= 1024 * 2304) return;
    int f = i % 2304, p = i / 2304;
    int c = f / 9, r = f % 9, dy = r / 3, dx = r % 3;
    int y0 = p / 32, x0 = p % 32;
    int yy = y0 + dy - 1, xx = x0 + dx - 1;
    float v = (yy >= 0 && yy < 32 && xx >= 0 && xx < 32)
                ? t2[(size_t)(yy * 32 + xx) * 256 + c] : 0.f;
    col[i] = (_Float16)v;
}

// [1024][256] -> [256][1024]  (NHWC -> NCHW final output)
__global__ void k_transpose_out(const float* __restrict__ in, float* __restrict__ out) {
    int i = blockIdx.x * blockDim.x + threadIdx.x;
    if (i >= 1024 * 256) return;
    int o = i % 256, p = i / 256;
    out[(size_t)o * 1024 + p] = in[i];
}

// ---------------------------------------------------------------------------
// Host orchestration
// ---------------------------------------------------------------------------
extern "C" void kernel_launch(void* const* d_in, const int* in_sizes, int n_in,
                              void* d_out, int out_size, void* d_ws, size_t ws_size,
                              hipStream_t stream) {
    (void)in_sizes; (void)n_in; (void)out_size; (void)ws_size;

    const float* x       = (const float*)d_in[0];
    const float* patch_w = (const float*)d_in[1];
    const float* patch_b = (const float*)d_in[2];
    const float* pos     = (const float*)d_in[3];
    const float* ln1_w   = (const float*)d_in[4];
    const float* ln1_b   = (const float*)d_in[5];
    const float* qkv_w   = (const float*)d_in[6];
    const float* qkv_b   = (const float*)d_in[7];
    const float* proj_w  = (const float*)d_in[8];
    const float* proj_b  = (const float*)d_in[9];
    const float* ln2_w   = (const float*)d_in[10];
    const float* ln2_b   = (const float*)d_in[11];
    const float* mlp1_w  = (const float*)d_in[12];
    const float* mlp1_b  = (const float*)d_in[13];
    const float* mlp2_w  = (const float*)d_in[14];
    const float* mlp2_b  = (const float*)d_in[15];
    const float* rhw     = (const float*)d_in[16];
    const float* rww     = (const float*)d_in[17];
    const float* rhg     = (const float*)d_in[18];
    const float* rwg     = (const float*)d_in[19];
    const float* neck_w1 = (const float*)d_in[20];
    const float* nln1_w  = (const float*)d_in[21];
    const float* nln1_b  = (const float*)d_in[22];
    const float* neck_w2 = (const float*)d_in[23];
    const float* nln2_w  = (const float*)d_in[24];
    const float* nln2_b  = (const float*)d_in[25];
    float* out = (float*)d_out;

    char* base = (char*)d_ws;
    size_t off = 0;
    auto alloc = [&](size_t bytes) -> void* {
        off = (off + 255) & ~(size_t)255;
        void* p = base + off;
        off += bytes;
        return p;
    };

    // All B operands stored transposed [N][K] for the GEMM
    _Float16* Wpatch = (_Float16*)alloc((size_t)768 * 768 * 2);        // [768][768]
    _Float16* Wqkv   = (_Float16*)alloc((size_t)12 * 2304 * 768 * 2);  // [2304][768]/layer
    _Float16* Wproj  = (_Float16*)alloc((size_t)12 * 768 * 768 * 2);   // [768][768]/layer
    _Float16* Wm1    = (_Float16*)alloc((size_t)12 * 3072 * 768 * 2);  // [3072][768]/layer
    _Float16* Wm2    = (_Float16*)alloc((size_t)12 * 768 * 3072 * 2);  // [768][3072]/layer
    _Float16* Wn1    = (_Float16*)alloc((size_t)256 * 768 * 2);        // [256][768]
    _Float16* Wn2    = (_Float16*)alloc((size_t)256 * 2304 * 2);       // [256][2304]
    _Float16* Xp     = (_Float16*)alloc((size_t)1024 * 768 * 2);
    float*    h      = (float*)alloc((size_t)1024 * 768 * 4);
    float*    yln    = (float*)alloc((size_t)1024 * 768 * 4);
    _Float16* yln_h  = (_Float16*)alloc((size_t)1024 * 768 * 2);
    float*    qkvo   = (float*)alloc((size_t)1024 * 2304 * 4);
    _Float16* qh     = (_Float16*)alloc((size_t)108 * 196 * 64 * 2);
    _Float16* kh     = (_Float16*)alloc((size_t)108 * 196 * 64 * 2);
    _Float16* vTh    = (_Float16*)alloc((size_t)108 * 196 * 64 * 2);
    float*    attn   = (float*)alloc((size_t)12 * 1024 * 1024 * 4);
    _Float16* attn_h = (_Float16*)alloc((size_t)12 * 1024 * 1024 * 2);
    float*    relh   = (float*)alloc((size_t)12 * 1024 * 32 * 4);
    float*    relw   = (float*)alloc((size_t)12 * 1024 * 32 * 4);
    float*    aout   = (float*)alloc((size_t)1024 * 768 * 4);
    _Float16* aout_h = (_Float16*)alloc((size_t)1024 * 768 * 2);
    float*    wout   = (float*)alloc((size_t)108 * 196 * 64 * 4);
    _Float16* mlp_h  = (_Float16*)alloc((size_t)1024 * 3072 * 2);
    _Float16* hh     = (_Float16*)alloc((size_t)1024 * 768 * 2);
    float*    t1     = (float*)alloc((size_t)1024 * 256 * 4);
    float*    t2     = (float*)alloc((size_t)1024 * 256 * 4);
    _Float16* colh   = (_Float16*)alloc((size_t)1024 * 2304 * 2);
    float*    t3     = (float*)alloc((size_t)1024 * 256 * 4);
    float*    t4     = (float*)alloc((size_t)1024 * 256 * 4);

    auto gemm = [&](const _Float16* A, int lda, size_t sA,
                    const _Float16* Bt, int ldb, size_t sB,
                    float* C, int ldc, size_t sC,
                    _Float16* Ch, int ldch, size_t sCh,
                    const float* bias, const float* res, int ldres, size_t sRes,
                    int M, int N, int K, int batch, int flags) {
        dim3 g((N + 63) / 64, (M + 63) / 64, batch);
        k_gemm<<<g, 128, 0, stream>>>(A, lda, sA, Bt, ldb, sB, C, ldc, sC,
                                      Ch, ldch, sCh, bias, res, ldres, sRes,
                                      M, N, K, flags);
    };
    auto cvt = [&](const float* s, _Float16* d, int n) {
        k_cvt<<<(n + 255) / 256, 256, 0, stream>>>(s, d, n);
    };
    auto cvt_t = [&](const float* s, _Float16* d, int R, int C, int batch) {
        dim3 g((R * C + 255) / 256, batch);
        k_cvt_t<<<g, 256, 0, stream>>>(s, d, R, C);
    };

    // --- weight conversion to f16, B-transposed layout (re-done every call) ---
    cvt(patch_w, Wpatch, 768 * 768);              // already [O][K] == [N][K]
    cvt_t(qkv_w,  Wqkv, 768, 2304, 12);           // -> [2304][768] per layer
    cvt_t(proj_w, Wproj, 768, 768, 12);
    cvt_t(mlp1_w, Wm1,   768, 3072, 12);          // -> [3072][768]
    cvt_t(mlp2_w, Wm2,   3072, 768, 12);          // -> [768][3072]
    cvt(neck_w1, Wn1, 256 * 768);                 // [256][768]
    cvt(neck_w2, Wn2, 256 * 2304);                // [256][2304]

    // --- patch embed: im2col + GEMM, pos_embed added as residual ---
    k_im2col_patch<<<(1024 * 768 + 255) / 256, 256, 0, stream>>>(x, Xp);
    gemm(Xp, 768, 0, Wpatch, 768, 0, h, 768, 0, nullptr, 0, 0,
         patch_b, pos, 768, 0, 1024, 768, 768, 1, 0);

    const float scale = 0.125f;   // 64^-0.5
    int wi = 0, gi = 0;
    for (int i = 0; i < 12; ++i) {
        bool isg = (i == 2 || i == 5 || i == 8 || i == 11);

        k_layernorm<<<1024, 256, 0, stream>>>(h, ln1_w + (size_t)i * 768,
                                              ln1_b + (size_t)i * 768, yln, yln_h, 768);
        gemm(yln_h, 768, 0, Wqkv + (size_t)i * 2304 * 768, 768, 0,
             qkvo, 2304, 0, nullptr, 0, 0,
             qkv_b + (size_t)i * 2304, nullptr, 0, 0, 1024, 2304, 768, 1, 0);

        if (isg) {
            k_extract_global<<<(12 * 1024 * 64 + 255) / 256, 256, 0, stream>>>(
                qkvo, qh, kh, vTh, scale);
            // attn = q @ k^T  (batch 12, M=N=1024, K=64); Bt = k itself
            gemm(qh, 64, (size_t)1024 * 64, kh, 64, (size_t)1024 * 64,
                 attn, 1024, (size_t)1024 * 1024, nullptr, 0, 0,
                 nullptr, nullptr, 0, 0, 1024, 1024, 64, 12, 0);
            dim3 gr((32 * 32 * 32 + 255) / 256, 12);
            k_relpos<<<gr, 256, 0, stream>>>(qh, rhg + (size_t)gi * 63 * 64, relh, 32, 32, 0, 8.0f);
            k_relpos<<<gr, 256, 0, stream>>>(qh, rwg + (size_t)gi * 63 * 64, relw, 32, 32, 1, 8.0f);
            k_rel_softmax<<<dim3(1024, 12), 256, 0, stream>>>(attn, attn_h, relh, relw, 32, 32);
            // out = attn @ v; Bt = v^T [64][1024]; heads concat via strided C
            gemm(attn_h, 1024, (size_t)1024 * 1024, vTh, 1024, (size_t)64 * 1024,
                 aout, 768, 64, aout_h, 768, 64,
                 nullptr, nullptr, 0, 0, 1024, 64, 1024, 12, 0);
            gi++;
        } else {
            k_extract_window<<<(108 * 196 * 64 + 255) / 256, 256, 0, stream>>>(
                qkvo, qkv_b + (size_t)i * 2304, qh, kh, vTh, scale);
            gemm(qh, 64, (size_t)196 * 64, kh, 64, (size_t)196 * 64,
                 attn, 196, (size_t)196 * 196, nullptr, 0, 0,
                 nullptr, nullptr, 0, 0, 196, 196, 64, 108, 0);
            dim3 gr((14 * 14 * 14 + 255) / 256, 108);
            k_relpos<<<gr, 256, 0, stream>>>(qh, rhw + (size_t)wi * 27 * 64, relh, 14, 14, 0, 8.0f);
            k_relpos<<<gr, 256, 0, stream>>>(qh, rww + (size_t)wi * 27 * 64, relw, 14, 14, 1, 8.0f);
            k_rel_softmax<<<dim3(196, 108), 256, 0, stream>>>(attn, attn_h, relh, relw, 14, 14);
            gemm(attn_h, 196, (size_t)196 * 196, vTh, 196, (size_t)64 * 196,
                 wout, 64, (size_t)196 * 64, nullptr, 0, 0,
                 nullptr, nullptr, 0, 0, 196, 64, 196, 108, 0);
            k_win_scatter<<<(108 * 196 * 64 + 255) / 256, 256, 0, stream>>>(wout, aout, aout_h);
            wi++;
        }

        // proj + residual into h
        gemm(aout_h, 768, 0, Wproj + (size_t)i * 768 * 768, 768, 0,
             h, 768, 0, nullptr, 0, 0,
             proj_b + (size_t)i * 768, h, 768, 0, 1024, 768, 768, 1, 0);

        // MLP
        k_layernorm<<<1024, 256, 0, stream>>>(h, ln2_w + (size_t)i * 768,
                                              ln2_b + (size_t)i * 768, yln, yln_h, 768);
        gemm(yln_h, 768, 0, Wm1 + (size_t)i * 3072 * 768, 768, 0,
             nullptr, 0, 0, mlp_h, 3072, 0,
             mlp1_b + (size_t)i * 3072, nullptr, 0, 0, 1024, 3072, 768, 1, GF_GELU);
        gemm(mlp_h, 3072, 0, Wm2 + (size_t)i * 768 * 3072, 3072, 0,
             h, 768, 0, nullptr, 0, 0,
             mlp2_b + (size_t)i * 768, h, 768, 0, 1024, 768, 3072, 1, 0);
    }

    // --- neck ---
    cvt(h, hh, 1024 * 768);
    gemm(hh, 768, 0, Wn1, 768, 0, t1, 256, 0, nullptr, 0, 0,
         nullptr, nullptr, 0, 0, 1024, 256, 768, 1, 0);
    k_layernorm<<<1024, 256, 0, stream>>>(t1, nln1_w, nln1_b, t2, nullptr, 256);
    k_im2col_3x3<<<(1024 * 2304 + 255) / 256, 256, 0, stream>>>(t2, colh);
    gemm(colh, 2304, 0, Wn2, 2304, 0, t3, 256, 0, nullptr, 0, 0,
         nullptr, nullptr, 0, 0, 1024, 256, 2304, 1, 0);
    k_layernorm<<<1024, 256, 0, stream>>>(t3, nln2_w, nln2_b, t4, nullptr, 256);
    k_transpose_out<<<(1024 * 256 + 255) / 256, 256, 0, stream>>>(t4, out);
}
